// HierMPNEncoder_4252017623665
// MI455X (gfx1250) — matrix operand order repaired and checked by reading, not compiled, not run on hardware
//
#include <hip/hip_runtime.h>
#include <hip/hip_bf16.h>
#include <math.h>

typedef __attribute__((ext_vector_type(16))) _Float16 v16h;
typedef __attribute__((ext_vector_type(8)))  float    v8f;

#define HDIM 256

__device__ __forceinline__ float sigmf(float x) { return 1.0f / (1.0f + expf(-x)); }
__device__ __forceinline__ float4 ld4(const float* p) { return *(const float4*)p; }
__device__ __forceinline__ void st4(float* p, float4 v) { *(float4*)p = v; }
__device__ __forceinline__ float4 add4(float4 a, float4 b) {
    return make_float4(a.x + b.x, a.y + b.y, a.z + b.z, a.w + b.w);
}

// ---------------------------------------------------------------------------
// Pre-pack fp32 weight slab W[row0 .. row0+K) x 256 into the wave32 WMMA
// B-fragment order this file uses for v_wmma_f32_16x16x32_f16 (contiguous k, matching A in k_gemm):
//   lane L holds column n = 16*nt + (L&15), K rows 32*ki + 16*(L>>4) + j, j=0..15
// ---------------------------------------------------------------------------
__global__ __launch_bounds__(256) void k_pack_w(const float* __restrict__ W, int row0, int K,
                                                _Float16* __restrict__ Wp)
{
    int gid = blockIdx.x * 256 + threadIdx.x;
    if (gid >= K * HDIM) return;
    int k = gid >> 8;
    int n = gid & 255;
    int ki = k >> 5, kk = k & 31;
    int half = kk >> 4, j = kk & 15;
    int lane = (half << 4) | (n & 15);
    int nt = n >> 4;
    Wp[(((size_t)(ki * 16 + nt) * 32 + lane) << 4) + j] =
        (_Float16)W[(size_t)(row0 + k) * HDIM + n];
}

// ---------------------------------------------------------------------------
// C[M x 256] = act( A[M x K] * Wpacked + bias )
// 128 threads = 4 waves. Block: 32-row strip; wave w: columns [64w, 64w+64).
// Each wave holds 2 A-fragments (rows 0-15, 16-31) x 4 N-tiles = 8 WMMA accums,
// so each B fragment load feeds two v_wmma ops. A rows are clamped (not
// predicated): OOB rows only feed C rows that are never stored.
// act: 0 = none, 1 = bias+relu, 2 = bias+tanh ; zero_row0: force row 0 -> 0
// ---------------------------------------------------------------------------
__global__ __launch_bounds__(128) void k_gemm(const float* __restrict__ A,
                                              const _Float16* __restrict__ Wp,
                                              const float* __restrict__ bias,
                                              float* __restrict__ C,
                                              int M, int K, int act, int zero_row0)
{
    __shared__ float As[32 * 32];
    const int tid  = threadIdx.x;
    const int lane = tid & 31, wave = tid >> 5;
    const int mloc = lane & 15, half = lane >> 4;
    const int row0 = blockIdx.x * 32;

    // A staging: thread -> (row 0..31, col8 0/8/16/24), unconditional b128 pairs
    const int ar = tid >> 2;
    const int ac = (tid & 3) << 3;
    int agr = row0 + ar;
    if (agr > M - 1) agr = M - 1;
    const float* Aptr = A + (size_t)agr * K + ac;

    v8f acc[2][4];
#pragma unroll
    for (int s = 0; s < 2; ++s)
#pragma unroll
        for (int t = 0; t < 4; ++t)
#pragma unroll
            for (int v = 0; v < 8; ++v) acc[s][t][v] = 0.0f;

    for (int k0 = 0; k0 < K; k0 += 32) {
        float4 v0 = ld4(Aptr + k0);
        float4 v1 = ld4(Aptr + k0 + 4);
        __syncthreads();
        st4(As + ar * 32 + ac, v0);
        st4(As + ar * 32 + ac + 4, v1);
        __syncthreads();

        v16h a0, a1;
#pragma unroll
        for (int i = 0; i < 8; ++i) {
            a0[i]     = (_Float16)As[mloc * 32 + half * 16 + i];
            a0[8 + i] = (_Float16)As[mloc * 32 + half * 16 + 8 + i];
            a1[i]     = (_Float16)As[(16 + mloc) * 32 + half * 16 + i];
            a1[8 + i] = (_Float16)As[(16 + mloc) * 32 + half * 16 + 8 + i];
        }

        const int ki = k0 >> 5;
        v16h b[4];
#pragma unroll
        for (int t = 0; t < 4; ++t)
            b[t] = *(const v16h*)(Wp +
                   (((size_t)(ki * 16 + (wave << 2) + t) * 32 + lane) << 4));
#pragma unroll
        for (int t = 0; t < 4; ++t) {
            acc[0][t] = __builtin_amdgcn_wmma_f32_16x16x32_f16(
                false, a0, false, b[t], (short)0, acc[0][t], false, false);
            acc[1][t] = __builtin_amdgcn_wmma_f32_16x16x32_f16(
                false, a1, false, b[t], (short)0, acc[1][t], false, false);
        }
    }

#pragma unroll
    for (int s = 0; s < 2; ++s) {
#pragma unroll
        for (int t = 0; t < 4; ++t) {
            const int ncol = (wave << 6) + (t << 4) + mloc;
#pragma unroll
            for (int v = 0; v < 8; ++v) {
                int gr = row0 + s * 16 + half * 8 + v;
                if (gr < M) {
                    float val = acc[s][t][v];
                    if (act == 1)      val = fmaxf(val + bias[ncol], 0.0f);
                    else if (act == 2) val = tanhf(val + bias[ncol]);
                    if (zero_row0 && gr == 0) val = 0.0f;
                    C[(size_t)gr * HDIM + ncol] = val;
                }
            }
        }
    }
}

// ---------------------------------------------------------------------------
// Row-wise kernels: 256 threads = 4 rows x 64 lanes, each lane owns 4 columns
// (float4) -> one b128 per access instead of four b32.
// ---------------------------------------------------------------------------

// graph-level GRU input projections: fmess is concat of one-hots -> row gathers
__global__ __launch_bounds__(256) void k_pre_g(const int* __restrict__ fmess,
        const int* __restrict__ fnode_atom,
        const float* __restrict__ Wz, const float* __restrict__ Wzb,
        const float* __restrict__ Wr,
        const float* __restrict__ Wh, const float* __restrict__ Whb,
        float* __restrict__ fz, float* __restrict__ fr, float* __restrict__ fh, int M)
{
    const int tid = threadIdx.x;
    const int m = blockIdx.x * 4 + (tid >> 6);
    if (m >= M) return;
    const int j = (tid & 63) << 2;
    const int src  = fmess[m * 4 + 0];
    const int bond = fmess[m * 4 + 2];
    const int pos  = fmess[m * 4 + 3];
    const int a = fnode_atom[src];
    const size_t o = (size_t)m * HDIM + j;
    const size_t oa = (size_t)a * HDIM + j;
    const size_t ob = (size_t)(40 + bond) * HDIM + j;
    const size_t op = (size_t)(44 + pos) * HDIM + j;
    st4(fz + o, add4(add4(ld4(Wz + oa), ld4(Wz + ob)), add4(ld4(Wz + op), ld4(Wzb + j))));
    st4(fr + o, add4(add4(ld4(Wr + oa), ld4(Wr + ob)), ld4(Wr + op)));
    st4(fh + o, add4(add4(ld4(Wh + oa), ld4(Wh + ob)), add4(ld4(Wh + op), ld4(Whb + j))));
}

// tree-level GRU input projections: dense part precomputed (hnW*), pos one-hot
__global__ __launch_bounds__(256) void k_pre_t(const int* __restrict__ fmess,
        const float* __restrict__ hnWz, const float* __restrict__ hnWr,
        const float* __restrict__ hnWh,
        const float* __restrict__ Wz, const float* __restrict__ Wzb,
        const float* __restrict__ Wr,
        const float* __restrict__ Wh, const float* __restrict__ Whb,
        float* __restrict__ fz, float* __restrict__ fr, float* __restrict__ fh, int M)
{
    const int tid = threadIdx.x;
    const int m = blockIdx.x * 4 + (tid >> 6);
    if (m >= M) return;
    const int j = (tid & 63) << 2;
    const int src = fmess[m * 3 + 0];
    const int pos = fmess[m * 3 + 2];
    const size_t o  = (size_t)m * HDIM + j;
    const size_t so = (size_t)src * HDIM + j;
    const size_t op = (size_t)(HDIM + pos) * HDIM + j;
    st4(fz + o, add4(ld4(hnWz + so), add4(ld4(Wz + op), ld4(Wzb + j))));
    st4(fr + o, add4(ld4(hnWr + so), ld4(Wr + op)));
    st4(fh + o, add4(ld4(hnWh + so), add4(ld4(Wh + op), ld4(Whb + j))));
}

// out[m] = sum_k src[graph[r,k]]  with r = rowidx ? rowidx[m] : m
__global__ __launch_bounds__(256) void k_gather_sum(const float* __restrict__ src,
        const int* __restrict__ graph, const int* __restrict__ rowidx, int nb,
        float* __restrict__ out, int M)
{
    const int tid = threadIdx.x;
    const int m = blockIdx.x * 4 + (tid >> 6);
    if (m >= M) return;
    const int j = (tid & 63) << 2;
    const int r = rowidx ? rowidx[m] : m;
    float4 s = make_float4(0.f, 0.f, 0.f, 0.f);
    for (int k = 0; k < nb; ++k) {
        int b = graph[(size_t)r * nb + k];
        s = add4(s, ld4(src + (size_t)b * HDIM + j));
    }
    st4(out + (size_t)m * HDIM + j, s);
}

// sum_gh[m] = sum_k sigmoid(fr[m] + hU[b] + Urb) * h[b],  b = bgraph[m,k]
__global__ __launch_bounds__(256) void k_sum_gh(const float* __restrict__ h,
        const float* __restrict__ hU, const float* __restrict__ fr,
        const float* __restrict__ Urb, const int* __restrict__ bgraph, int nb,
        float* __restrict__ out, int M)
{
    const int tid = threadIdx.x;
    const int m = blockIdx.x * 4 + (tid >> 6);
    if (m >= M) return;
    const int j = (tid & 63) << 2;
    const float4 frv = ld4(fr + (size_t)m * HDIM + j);
    const float4 bb  = ld4(Urb + j);
    float4 s = make_float4(0.f, 0.f, 0.f, 0.f);
    for (int k = 0; k < nb; ++k) {
        int b = bgraph[(size_t)m * nb + k];
        float4 hv = ld4(h  + (size_t)b * HDIM + j);
        float4 uv = ld4(hU + (size_t)b * HDIM + j);
        s.x += sigmf(frv.x + uv.x + bb.x) * hv.x;
        s.y += sigmf(frv.y + uv.y + bb.y) * hv.y;
        s.z += sigmf(frv.z + uv.z + bb.z) * hv.z;
        s.w += sigmf(frv.w + uv.w + bb.w) * hv.w;
    }
    st4(out + (size_t)m * HDIM + j, s);
}

// h = mask * ((1-z)*sum_h + z*tanh(fh+hpart)),  z = sigmoid(fz+zpart)
__global__ __launch_bounds__(256) void k_update(const float* __restrict__ fz,
        const float* __restrict__ zpart, const float* __restrict__ fh,
        const float* __restrict__ hpart, const float* __restrict__ sum_h,
        float* __restrict__ h, int M)
{
    const int tid = threadIdx.x;
    const int m = blockIdx.x * 4 + (tid >> 6);
    if (m >= M) return;
    const int j = (tid & 63) << 2;
    const size_t o = (size_t)m * HDIM + j;
    const float4 a = ld4(fz + o), b = ld4(zpart + o);
    const float4 c = ld4(fh + o), d = ld4(hpart + o);
    const float4 sh = ld4(sum_h + o);
    float4 r;
    {
        float z = sigmf(a.x + b.x), p = tanhf(c.x + d.x);
        r.x = (1.0f - z) * sh.x + z * p;
        z = sigmf(a.y + b.y); p = tanhf(c.y + d.y);
        r.y = (1.0f - z) * sh.y + z * p;
        z = sigmf(a.z + b.z); p = tanhf(c.z + d.z);
        r.z = (1.0f - z) * sh.z + z * p;
        z = sigmf(a.w + b.w); p = tanhf(c.w + d.w);
        r.w = (1.0f - z) * sh.w + z * p;
    }
    if (m == 0) r = make_float4(0.f, 0.f, 0.f, 0.f);
    st4(h + o, r);
}

// hatom[v] = relu(Wo[atom_onehot_row] + nei@Wo[40:] + b), row 0 zeroed
__global__ __launch_bounds__(256) void k_out_g(const int* __restrict__ fnode_atom,
        const float* __restrict__ Wo, const float* __restrict__ Wob,
        const float* __restrict__ neiWo, float* __restrict__ out, int M)
{
    const int tid = threadIdx.x;
    const int v = blockIdx.x * 4 + (tid >> 6);
    if (v >= M) return;
    const int j = (tid & 63) << 2;
    const int a = fnode_atom[v];
    float4 s = add4(add4(ld4(Wo + (size_t)a * HDIM + j),
                         ld4(neiWo + (size_t)v * HDIM + j)), ld4(Wob + j));
    s.x = fmaxf(s.x, 0.f); s.y = fmaxf(s.y, 0.f);
    s.z = fmaxf(s.z, 0.f); s.w = fmaxf(s.w, 0.f);
    if (v == 0) s = make_float4(0.f, 0.f, 0.f, 0.f);
    st4(out + (size_t)v * HDIM + j, s);
}

// X[m] = [ A[idxA ? idxA[m*strideA+offA] : m] | B[m] ]   (512 wide)
__global__ __launch_bounds__(256) void k_concat(const float* __restrict__ A,
        const int* __restrict__ idxA, int strideA, int offA,
        const float* __restrict__ B, float* __restrict__ X, int M)
{
    const int tid = threadIdx.x;
    const int m = blockIdx.x * 4 + (tid >> 6);
    if (m >= M) return;
    const int j = (tid & 63) << 2;
    const int ia = idxA ? idxA[(size_t)m * strideA + offA] : m;
    st4(X + (size_t)m * 512 + j,       ld4(A + (size_t)ia * HDIM + j));
    st4(X + (size_t)m * 512 + 256 + j, ld4(B + (size_t)m * HDIM + j));
}

// ---------------------------------------------------------------------------
extern "C" void kernel_launch(void* const* d_in, const int* in_sizes, int n_in,
                              void* d_out, int out_size, void* d_ws, size_t ws_size,
                              hipStream_t stream)
{
    (void)in_sizes; (void)n_in; (void)out_size; (void)ws_size;
    const int NT = 8000, MT = 16000, NA = 25000, MA = 50000;
    const int NB = 6, NC = 12, NROOT = 256;

    // ---- inputs (setup_inputs() order) ----
    const int* t_fnode  = (const int*)d_in[0];
    const int* t_fmess  = (const int*)d_in[1];
    const int* t_agraph = (const int*)d_in[2];
    const int* t_bgraph = (const int*)d_in[3];
    const int* t_cgraph = (const int*)d_in[4];
    const int* g_fnode  = (const int*)d_in[5];
    const int* g_fmess  = (const int*)d_in[6];
    const int* g_agraph = (const int*)d_in[7];
    const int* g_bgraph = (const int*)d_in[8];
    const int* roots    = (const int*)d_in[9];
    const float* E_c    = (const float*)d_in[10];
    const float* E_i    = (const float*)d_in[11];
    const float* Wc_b   = (const float*)d_in[13];
    const float* Wi_b   = (const float*)d_in[15];
    const float* Wroot_b= (const float*)d_in[17];
    const float* t_Wo_b = (const float*)d_in[19];
    const float* t_Wz_w = (const float*)d_in[20];
    const float* t_Wz_b = (const float*)d_in[21];
    const float* t_Wr_w = (const float*)d_in[22];
    const float* t_Ur_b = (const float*)d_in[24];
    const float* t_Wh_w = (const float*)d_in[25];
    const float* t_Wh_b = (const float*)d_in[26];
    const float* i_Wo_b = (const float*)d_in[28];
    const float* i_Wz_w = (const float*)d_in[29];
    const float* i_Wz_b = (const float*)d_in[30];
    const float* i_Wr_w = (const float*)d_in[31];
    const float* i_Ur_b = (const float*)d_in[33];
    const float* i_Wh_w = (const float*)d_in[34];
    const float* i_Wh_b = (const float*)d_in[35];
    const float* g_Wo_w = (const float*)d_in[36];
    const float* g_Wo_b = (const float*)d_in[37];
    const float* g_Wz_w = (const float*)d_in[38];
    const float* g_Wz_b = (const float*)d_in[39];
    const float* g_Wr_w = (const float*)d_in[40];
    const float* g_Ur_b = (const float*)d_in[42];
    const float* g_Wh_w = (const float*)d_in[43];
    const float* g_Wh_b = (const float*)d_in[44];

    // ---- outputs ----
    float* out      = (float*)d_out;
    float* hroot_o  = out;
    float* htree_o  = out + (size_t)NROOT * HDIM;
    float* hinter_o = htree_o + (size_t)NT * HDIM;
    float* hatom_o  = hinter_o + (size_t)NT * HDIM;

    // ---- workspace carve ----
    char* wsb = (char*)d_ws;
    size_t off = 0;
    auto carve = [&](size_t bytes) -> void* {
        void* p = wsb + off;
        off += (bytes + 255) & ~(size_t)255;
        return p;
    };
    enum { PWi = 0, PWc, PWroot, PgUr, PgWzh, PgWhh, PgWo,
           PiUr, PiWzh, PiWhh, PiWz0, PiWr0, PiWh0, PiWo,
           PtUr, PtWzh, PtWhh, PtWz0, PtWr0, PtWh0, PtWo, NPACK };
    _Float16* P[NPACK];
    for (int i = 0; i < NPACK; ++i) P[i] = (_Float16*)carve(512 * 256 * 2);

    const size_t ROWB = (size_t)HDIM * 4;
    float* fz     = (float*)carve((size_t)MA * ROWB);
    float* fr     = (float*)carve((size_t)MA * ROWB);
    float* fh     = (float*)carve((size_t)MA * ROWB);
    float* h      = (float*)carve((size_t)MA * ROWB);
    float* sum_h  = (float*)carve((size_t)MA * ROWB);
    float* hU     = (float*)carve((size_t)MA * ROWB);   // reused as hpart / neiWo
    float* sum_gh = (float*)carve((size_t)MA * ROWB);
    float* zpart  = (float*)carve((size_t)MA * ROWB);
    float* hn     = (float*)carve((size_t)NT * ROWB);
    float* hn2    = (float*)carve((size_t)NT * ROWB);
    float* cg     = (float*)carve((size_t)NT * ROWB);
    float* hnWz   = (float*)carve((size_t)NT * ROWB);
    float* hnWr   = (float*)carve((size_t)NT * ROWB);
    float* hnWh   = (float*)carve((size_t)NT * ROWB);
    float* X      = (float*)carve((size_t)NT * 512 * 4);
    float* nei_r  = (float*)carve((size_t)NROOT * ROWB);

    auto pack = [&](int in_idx, int row0, int K, _Float16* dst) {
        int n = K * HDIM;
        k_pack_w<<<dim3((n + 255) / 256), dim3(256), 0, stream>>>(
            (const float*)d_in[in_idx], row0, K, dst);
    };
    auto gemm = [&](const float* A, int M, int K, const _Float16* Wp,
                    const float* bias, int act, int zr0, float* C) {
        k_gemm<<<dim3((M + 31) / 32), dim3(128), 0, stream>>>(A, Wp, bias, C, M, K, act, zr0);
    };
    auto rows4 = [](int r) { return dim3((r + 3) / 4); };

    // ---- pack all weights once ----
    pack(14, 0, 512, P[PWi]);
    pack(12, 0, 512, P[PWc]);
    pack(16, 0, 512, P[PWroot]);
    pack(41, 0, 256, P[PgUr]);   pack(38, 64, 256, P[PgWzh]);
    pack(43, 64, 256, P[PgWhh]); pack(36, 40, 256, P[PgWo]);
    pack(32, 0, 256, P[PiUr]);   pack(29, 276, 256, P[PiWzh]);
    pack(34, 276, 256, P[PiWhh]);
    pack(29, 0, 256, P[PiWz0]);  pack(31, 0, 256, P[PiWr0]);
    pack(34, 0, 256, P[PiWh0]);  pack(27, 0, 512, P[PiWo]);
    pack(23, 0, 256, P[PtUr]);   pack(20, 276, 256, P[PtWzh]);
    pack(25, 276, 256, P[PtWhh]);
    pack(20, 0, 256, P[PtWz0]);  pack(22, 0, 256, P[PtWr0]);
    pack(25, 0, 256, P[PtWh0]);  pack(18, 0, 512, P[PtWo]);

    auto gru_steps = [&](int M, const int* bgraph,
                         const _Float16* pUr, const _Float16* pWzh, const _Float16* pWhh,
                         const float* Urb, int depth) {
        for (int d = 0; d < depth; ++d) {
            k_gather_sum<<<rows4(M), dim3(256), 0, stream>>>(h, bgraph, nullptr, NB, sum_h, M);
            gemm(h, M, 256, pUr, nullptr, 0, 0, hU);
            gemm(sum_h, M, 256, pWzh, nullptr, 0, 0, zpart);
            k_sum_gh<<<rows4(M), dim3(256), 0, stream>>>(h, hU, fr, Urb, bgraph, NB, sum_gh, M);
            gemm(sum_gh, M, 256, pWhh, nullptr, 0, 0, hU);    // hpart -> hU buffer
            k_update<<<rows4(M), dim3(256), 0, stream>>>(fz, zpart, fh, hU, sum_h, h, M);
        }
    };

    // ================= graph (atom) MPN =================
    hipMemsetAsync(h, 0, (size_t)MA * ROWB, stream);
    k_pre_g<<<rows4(MA), dim3(256), 0, stream>>>(g_fmess, g_fnode,
        g_Wz_w, g_Wz_b, g_Wr_w, g_Wh_w, g_Wh_b, fz, fr, fh, MA);
    gru_steps(MA, g_bgraph, P[PgUr], P[PgWzh], P[PgWhh], g_Ur_b, 5);
    float* nei = sum_h;     // reuse (GRU done)
    float* neiWo = hU;
    k_gather_sum<<<rows4(NA), dim3(256), 0, stream>>>(h, g_agraph, nullptr, NB, nei, NA);
    gemm(nei, NA, 256, P[PgWo], nullptr, 0, 0, neiWo);
    k_out_g<<<rows4(NA), dim3(256), 0, stream>>>(g_fnode, g_Wo_w, g_Wo_b, neiWo, hatom_o, NA);

    // ================= tree candidate vectors (Wi) =================
    k_gather_sum<<<rows4(NT), dim3(256), 0, stream>>>(hatom_o, t_cgraph, nullptr, NC, cg, NT);
    k_concat<<<rows4(NT), dim3(256), 0, stream>>>(E_i, t_fnode, 2, 1, cg, X, NT);
    gemm(X, NT, 512, P[PWi], Wi_b, 1, 0, hn);

    auto tree_mpn = [&](const float* hin,
                        const _Float16* pWz0, const _Float16* pWr0, const _Float16* pWh0,
                        const _Float16* pUr, const _Float16* pWzh, const _Float16* pWhh,
                        const _Float16* pWo,
                        const float* Wz_w, const float* Wz_b, const float* Wr_w,
                        const float* Wh_w, const float* Wh_b,
                        const float* Ur_b, const float* Wo_b, float* nh_out) {
        gemm(hin, NT, 256, pWz0, nullptr, 0, 0, hnWz);
        gemm(hin, NT, 256, pWr0, nullptr, 0, 0, hnWr);
        gemm(hin, NT, 256, pWh0, nullptr, 0, 0, hnWh);
        k_pre_t<<<rows4(MT), dim3(256), 0, stream>>>(t_fmess, hnWz, hnWr, hnWh,
            Wz_w, Wz_b, Wr_w, Wh_w, Wh_b, fz, fr, fh, MT);
        hipMemsetAsync(h, 0, (size_t)MT * ROWB, stream);
        gru_steps(MT, t_bgraph, pUr, pWzh, pWhh, Ur_b, 5);
        float* nt_ = sum_h; // reuse
        k_gather_sum<<<rows4(NT), dim3(256), 0, stream>>>(h, t_agraph, nullptr, NB, nt_, NT);
        k_concat<<<rows4(NT), dim3(256), 0, stream>>>(hin, nullptr, 0, 0, nt_, X, NT);
        gemm(X, NT, 512, pWo, Wo_b, 1, 1, nh_out);
    };

    // ================= inter MPN =================
    tree_mpn(hn, P[PiWz0], P[PiWr0], P[PiWh0], P[PiUr], P[PiWzh], P[PiWhh], P[PiWo],
             i_Wz_w, i_Wz_b, i_Wr_w, i_Wh_w, i_Wh_b, i_Ur_b, i_Wo_b, hinter_o);

    // ================= Wc layer =================
    k_concat<<<rows4(NT), dim3(256), 0, stream>>>(E_c, t_fnode, 2, 0, hinter_o, X, NT);
    gemm(X, NT, 512, P[PWc], Wc_b, 1, 0, hn2);

    // ================= tree MPN (keeps h = h_msg) =================
    tree_mpn(hn2, P[PtWz0], P[PtWr0], P[PtWh0], P[PtUr], P[PtWzh], P[PtWhh], P[PtWo],
             t_Wz_w, t_Wz_b, t_Wr_w, t_Wh_w, t_Wh_b, t_Ur_b, t_Wo_b, htree_o);

    // ================= root readout =================
    k_gather_sum<<<rows4(NROOT), dim3(256), 0, stream>>>(h, t_agraph, roots, NB, nei_r, NROOT);
    k_concat<<<rows4(NROOT), dim3(256), 0, stream>>>(hn2, roots, 1, 0, nei_r, X, NROOT);
    gemm(X, NROOT, 512, P[PWroot], Wroot_b, 2, 0, hroot_o);
}
